// ExtractorMLP_9620726743307
// MI455X (gfx1250) — matrix operand order repaired and checked by reading, not compiled, not run on hardware
//
#include <hip/hip_runtime.h>
#include <hip/hip_bf16.h>

// ---------------------------------------------------------------------------
// ExtractorMLP edge-attention:  out[e] = relu(relu([emb[col],emb[row]]W1+b1)W2+b2)W3+b3
// Fused per 16-edge tile with v_wmma_f32_16x16x32_f16; weights staged once per
// workgroup in LDS via double-buffered async-to-LDS copies (8x L2 traffic cut).
// ---------------------------------------------------------------------------

typedef __attribute__((ext_vector_type(16))) _Float16 v16h;
typedef __attribute__((ext_vector_type(8)))  _Float16 v8h;
typedef __attribute__((ext_vector_type(8)))  float    v8f;
typedef __attribute__((ext_vector_type(4)))  int      v4i;

#define AS_GLOBAL __attribute__((address_space(1)))
#define AS_LDS    __attribute__((address_space(3)))

#define HID    128
#define E_TOT  640000
#define D_IN   256      // 2*HID
#define D_H1   512
#define D_H2   128

#define W1P_HALVES (D_IN * D_H1)   // 131072 (256 KB as f16)
#define W2P_HALVES (D_H1 * D_H2)   // 65536  (128 KB as f16)

#define W1_CHUNK_HALVES 8192       // per-kc: 2 N-tiles x 8 K-chunks x 512 = 16 KB
#define W2_CHUNK_HALVES 4096       // per-kc: 8 N-tiles x 512            =  8 KB
#define STAGE_HALVES (W1_CHUNK_HALVES + W2_CHUNK_HALVES)   // 24 KB

#if defined(__has_builtin)
#if __has_builtin(__builtin_amdgcn_global_load_async_to_lds_b128) && \
    __has_builtin(__builtin_amdgcn_s_wait_asynccnt)
#define HAVE_ASYNC_LDS 1
#endif
#endif
#ifndef HAVE_ASYNC_LDS
#define HAVE_ASYNC_LDS 0
#endif

// ---------------------------------------------------------------------------
// Pre-pass: convert W1/W2 (f32 row-major, [K,N]) into f16 WMMA B-fragment
// order, arranged so each kc iteration's fragments are one contiguous block.
//   W1p: [t1(0..31)][c(0..7)][lane(0..31)][i(0..15)]  -> kc chunk = t1 in
//        {2kc, 2kc+1} = contiguous 16 KB at kc*8192 halves.
//   W2p: [c(0..15)][j(0..7)][lane][i]                 -> kc chunk = contiguous
//        8 KB at kc*4096 halves.
// Lane l = 16*h + n; half index i -> K = 32c + 8h + i (i<8) or 32c + 8h + 8 + i (i>=8); N = 16*tile + n.
// ---------------------------------------------------------------------------
__global__ void pack_weights_kernel(const float* __restrict__ W1,
                                    const float* __restrict__ W2,
                                    _Float16* __restrict__ W1p,
                                    _Float16* __restrict__ W2p) {
  int t = blockIdx.x * blockDim.x + threadIdx.x;
  if (t < W1P_HALVES) {
    int i  = t & 15;
    int l  = (t >> 4) & 31;
    int c  = (t >> 9) & 7;        // 8 K-chunks of 32 over D_IN=256
    int t1 = (t >> 12) & 31;      // 32 N-tiles of 16 over D_H1=512
    int n = l & 15, h = l >> 4;
    int K = 32 * c + 8 * h + (i & 7) + 2 * (i & 8);
    int N = 16 * t1 + n;
    W1p[t] = (_Float16)W1[K * D_H1 + N];
  } else if (t < W1P_HALVES + W2P_HALVES) {
    int u  = t - W1P_HALVES;
    int i  = u & 15;
    int l  = (u >> 4) & 31;
    int t2 = (u >> 9) & 7;        // 8 N-tiles of 16 over D_H2=128
    int c  = u >> 12;             // 16 K-chunks of 32 over D_H1=512
    int n = l & 15, h = l >> 4;
    int K = 32 * c + 8 * h + (i & 7) + 2 * (i & 8);
    int N = 16 * t2 + n;
    W2p[u] = (_Float16)W2[K * D_H2 + N];
  }
}

// ---------------------------------------------------------------------------
// Stage one kc's weight chunk (24 KB) into LDS, shared by all 8 waves.
// 256 threads x 6 x 16B. Async path: one GLOBAL_LOAD_ASYNC_TO_LDS_B128 per
// lane-call (512 B/wave-instruction), tracked by ASYNCcnt.
// ---------------------------------------------------------------------------
__device__ __forceinline__ void stage_weights(int kc, _Float16* dst,
                                              const _Float16* __restrict__ W1p,
                                              const _Float16* __restrict__ W2p,
                                              int tid) {
  const _Float16* s1 = W1p + kc * W1_CHUNK_HALVES;
  const _Float16* s2 = W2p + kc * W2_CHUNK_HALVES;
#if HAVE_ASYNC_LDS
#pragma unroll
  for (int q = 0; q < 4; ++q) {
    int off = (q * 256 + tid) * 8;   // halves; 16 B per transfer
    __builtin_amdgcn_global_load_async_to_lds_b128(
        (AS_GLOBAL v4i*)(s1 + off),
        (AS_LDS v4i*)(dst + off), 0, 0);
  }
#pragma unroll
  for (int q = 0; q < 2; ++q) {
    int off = (q * 256 + tid) * 8;
    __builtin_amdgcn_global_load_async_to_lds_b128(
        (AS_GLOBAL v4i*)(s2 + off),
        (AS_LDS v4i*)(dst + W1_CHUNK_HALVES + off), 0, 0);
  }
#else
  // sync fallback: VGPR round trip (still shares the chunk across 8 waves)
#pragma unroll
  for (int q = 0; q < 4; ++q) {
    int off = (q * 256 + tid) * 8;
    *reinterpret_cast<float4*>(dst + off) =
        *reinterpret_cast<const float4*>(s1 + off);
  }
#pragma unroll
  for (int q = 0; q < 2; ++q) {
    int off = (q * 256 + tid) * 8;
    *reinterpret_cast<float4*>(dst + W1_CHUNK_HALVES + off) =
        *reinterpret_cast<const float4*>(s2 + off);
  }
#endif
}

__device__ __forceinline__ void wait_stage() {
#if HAVE_ASYNC_LDS
  __builtin_amdgcn_s_wait_asynccnt(0);
#endif
  __syncthreads();
}

// ---------------------------------------------------------------------------
// Main fused kernel: 8 waves/block, each wave processes 16 edges end-to-end.
// ---------------------------------------------------------------------------
__global__ __launch_bounds__(256)
void extractor_mlp_kernel(const float* __restrict__ emb,
                          const int*   __restrict__ eidx,   // [2, E]
                          const float* __restrict__ b1,
                          const float* __restrict__ b2,
                          const float* __restrict__ W3,
                          const float* __restrict__ b3,
                          const _Float16* __restrict__ W1p,
                          const _Float16* __restrict__ W2p,
                          float* __restrict__ out) {
  // double-buffered weight stage (2 x 24 KB) + per-wave 16x32 h1 scratch (8 KB)
  __shared__ __align__(16) _Float16 stage[2][STAGE_HALVES];
  __shared__ __align__(16) _Float16 sh1[8 * 16 * 32];

  const int tid  = threadIdx.x;
  const int lane = tid & 31;
  const int wave = tid >> 5;
  const int e0   = (blockIdx.x * 8 + wave) * 16;  // 16-edge tile base
  const int m    = lane & 15;                     // A-row / C-column lane id
  const int h    = lane >> 4;                     // K-half select

  // kick off the kc=0 weight stage before doing anything else
  stage_weights(0, stage[0], W1p, W2p, tid);

  const int ci = eidx[e0 + m];           // col -> first 128 cols of f12
  const int ri = eidx[E_TOT + e0 + m];   // row -> last 128 cols of f12

  // ---- A1: f12 rows (16 edges x 256) as 8 f16 A-fragments, in registers ----
  // 16-bit A 16x32 layout: lane m holds row m; half index i -> K_local =
  // 8h + i (i<8) or 8h + 8 + i (i>=8).  4x float4 loads per fragment.
  v16h A1[8];
#pragma unroll
  for (int c = 0; c < 8; ++c) {
    const int kb = 32 * c;
    const float* src = (kb < HID) ? (emb + (size_t)ci * HID + kb)
                                  : (emb + (size_t)ri * HID + (kb - HID));
    const float* base = src + 8 * h;
    float4 f0 = *reinterpret_cast<const float4*>(base + 0);
    float4 f1 = *reinterpret_cast<const float4*>(base + 4);
    float4 f2 = *reinterpret_cast<const float4*>(base + 16);
    float4 f3 = *reinterpret_cast<const float4*>(base + 20);
    A1[c][0]  = (_Float16)f0.x; A1[c][1]  = (_Float16)f0.y;
    A1[c][2]  = (_Float16)f0.z; A1[c][3]  = (_Float16)f0.w;
    A1[c][4]  = (_Float16)f1.x; A1[c][5]  = (_Float16)f1.y;
    A1[c][6]  = (_Float16)f1.z; A1[c][7]  = (_Float16)f1.w;
    A1[c][8]  = (_Float16)f2.x; A1[c][9]  = (_Float16)f2.y;
    A1[c][10] = (_Float16)f2.z; A1[c][11] = (_Float16)f2.w;
    A1[c][12] = (_Float16)f3.x; A1[c][13] = (_Float16)f3.y;
    A1[c][14] = (_Float16)f3.z; A1[c][15] = (_Float16)f3.w;
  }

  // per-lane layer-3 constants (lane n=m of each 16-wide N-tile j)
  float w3v[8], b2v[8];
#pragma unroll
  for (int j = 0; j < 8; ++j) {
    w3v[j] = W3[16 * j + m];
    b2v[j] = b2[16 * j + m];
  }

  _Float16* shw = sh1 + wave * (16 * 32);

  v8f C2[8];
#pragma unroll
  for (int j = 0; j < 8; ++j) C2[j] = (v8f){};

  wait_stage();   // stage[0] ready

  // ---- fused layer1 -> layer2 over 16 chunks of 32 h1-columns ----
#pragma unroll 1
  for (int kc = 0; kc < 16; ++kc) {
    // prefetch next chunk into the other buffer (last read in kc-1, so the
    // barrier at the end of kc-1 guarantees nobody is still reading it)
    if (kc < 15) stage_weights(kc + 1, stage[(kc + 1) & 1], W1p, W2p, tid);

    const _Float16* sb = stage[kc & 1];
    const _Float16* sb2 = sb + W1_CHUNK_HALVES;

    v8f c1a = (v8f){};
    v8f c1b = (v8f){};
#pragma unroll
    for (int c = 0; c < 8; ++c) {
      v16h bA = *reinterpret_cast<const v16h*>(sb + (c)*512 + lane * 16);
      v16h bB = *reinterpret_cast<const v16h*>(sb + (8 + c) * 512 + lane * 16);
      c1a = __builtin_amdgcn_wmma_f32_16x16x32_f16(false, A1[c], false, bA,
                                                   (short)0, c1a, false, false);
      c1b = __builtin_amdgcn_wmma_f32_16x16x32_f16(false, A1[c], false, bB,
                                                   (short)0, c1b, false, false);
    }
    // bias + ReLU, spill 16x32 chunk to per-wave LDS as f16
    const float bva = b1[32 * kc + m];
    const float bvb = b1[32 * kc + 16 + m];
#pragma unroll
    for (int r = 0; r < 8; ++r) {
      const int row = r + 8 * h;                    // C layout: M = r + 8h, N = m
      float xa = fmaxf(c1a[r] + bva, 0.0f);
      float xb = fmaxf(c1b[r] + bvb, 0.0f);
      shw[row * 32 + m]      = (_Float16)xa;
      shw[row * 32 + 16 + m] = (_Float16)xb;
    }
    // reload as layer-2 A fragment (two ds_load_b128 per lane; same-wave LDS
    // ops are in-order, region is wave-private -> no barrier needed)
    const _Float16* ab = shw + m * 32 + 8 * h;
    v8h lo = *reinterpret_cast<const v8h*>(ab);        // K_local = 8h + 0..7
    v8h hi = *reinterpret_cast<const v8h*>(ab + 16);   // K_local = 8h + 16..23
    v16h A2;
#pragma unroll
    for (int i = 0; i < 8; ++i) { A2[i] = lo[i]; A2[8 + i] = hi[i]; }

#pragma unroll
    for (int j = 0; j < 8; ++j) {
      v16h bW = *reinterpret_cast<const v16h*>(sb2 + j * 512 + lane * 16);
      C2[j] = __builtin_amdgcn_wmma_f32_16x16x32_f16(false, A2, false, bW,
                                                     (short)0, C2[j], false, false);
    }

    // make this iteration's prefetch visible to everyone before kc+1 reads it
    wait_stage();
  }

  // ---- layer 3: relu(h2 + b2) . W3 + b3, butterfly-reduced per row ----
  const float b3s = b3[0];
#pragma unroll
  for (int r = 0; r < 8; ++r) {
    float s = 0.0f;
#pragma unroll
    for (int j = 0; j < 8; ++j) {
      float x = fmaxf(C2[j][r] + b2v[j], 0.0f);
      s = fmaf(x, w3v[j], s);
    }
    // sum the 16 N-lanes of this half (xor<16 never crosses the half boundary)
    s += __shfl_xor(s, 1);
    s += __shfl_xor(s, 2);
    s += __shfl_xor(s, 4);
    s += __shfl_xor(s, 8);
    if (m == 0) out[e0 + r + 8 * h] = s + b3s;
  }
}

// ---------------------------------------------------------------------------
extern "C" void kernel_launch(void* const* d_in, const int* in_sizes, int n_in,
                              void* d_out, int out_size, void* d_ws, size_t ws_size,
                              hipStream_t stream) {
  const float* emb  = (const float*)d_in[0];
  const int*   eidx = (const int*)d_in[1];
  // d_in[2] = batch (unused by the reference output)
  const float* W1 = (const float*)d_in[3];
  const float* b1 = (const float*)d_in[4];
  const float* W2 = (const float*)d_in[5];
  const float* b2 = (const float*)d_in[6];
  const float* W3 = (const float*)d_in[7];
  const float* b3 = (const float*)d_in[8];
  float* out = (float*)d_out;

  _Float16* W1p = (_Float16*)d_ws;               // 256 KB
  _Float16* W2p = W1p + W1P_HALVES;              // +128 KB

  const int packElems = W1P_HALVES + W2P_HALVES; // 196608
  pack_weights_kernel<<<(packElems + 255) / 256, 256, 0, stream>>>(W1, W2, W1p, W2p);

  const int blocks = E_TOT / (16 * 8);           // 5000 blocks, 128 edges each
  extractor_mlp_kernel<<<blocks, 256, 0, stream>>>(emb, eidx, b1, b2, W3, b3,
                                                   W1p, W2p, out);
}
